// ConformerBlock_56435870269589
// MI455X (gfx1250) — matrix operand order repaired and checked by reading, not compiled, not run on hardware
//
#include <hip/hip_runtime.h>
#include <cstdint>
#include <cstddef>

// ---------------------------------------------------------------------------
// CDNA5 (gfx1250) Conformer block. wave32, WMMA bf16 (16x16x32), f32 accum.
// GEMM tiles staged by the Tensor Data Mover (tensor_load_to_lds, TENSORcnt),
// double-buffered in LDS; WMMA operands are 2x ds_load_b128 / global_load_b128.
// ---------------------------------------------------------------------------

typedef __attribute__((ext_vector_type(16))) __bf16          v16bf;
typedef __attribute__((ext_vector_type(8)))  float           v8f;
typedef __attribute__((ext_vector_type(8)))  unsigned short  us8;
typedef __attribute__((ext_vector_type(16))) unsigned short  us16;
typedef __attribute__((ext_vector_type(4)))  unsigned int    u32x4;
typedef __attribute__((ext_vector_type(4)))  int             i32x4;
typedef __attribute__((ext_vector_type(8)))  int             i32x8;

#define DM   512
#define TT   1024
#define NBAT 8
#define NH   8
#define DKH  64

static __device__ __forceinline__ unsigned short f2bf(float f) {
  unsigned u = __builtin_bit_cast(unsigned, f);
  u += 0x7fffu + ((u >> 16) & 1u);            // round-to-nearest-even
  return (unsigned short)(u >> 16);
}
static __device__ __forceinline__ float bf2f(unsigned short u) {
  return __builtin_bit_cast(float, ((unsigned)u) << 16);
}
static __device__ __forceinline__ float sigm(float x) {
  return 1.f / (1.f + __expf(-x));
}
// Build a v16bf WMMA operand from two contiguous 16-byte chunks.
static __device__ __forceinline__ v16bf frag2(const unsigned short* p0,
                                              const unsigned short* p1) {
  us8 a = *(const us8*)p0;
  us8 b = *(const us8*)p1;
  us16 w = __builtin_shufflevector(a, b, 0,1,2,3,4,5,6,7,8,9,10,11,12,13,14,15);
  return __builtin_bit_cast(v16bf, w);
}
// Build a v16bf WMMA operand from one contiguous 32-byte chunk.
static __device__ __forceinline__ v16bf frag1(const unsigned short* p) {
  return __builtin_bit_cast(v16bf, *(const us16*)p);
}

// ---------------------------------------------------------------------------
// Tensor Data Mover: load a [rows x 32] bf16 tile (row stride = strideElems)
// from global into LDS at ldsByteAddr, padding each 64B row with 16B so the
// LDS row stride is 80B (bank-friendly, 16B aligned).
// D# layout per CDNA5 ISA.
// ---------------------------------------------------------------------------
static __device__ __forceinline__ void tdm_load_tile(
    const unsigned short* gptr, unsigned rows, unsigned strideElems,
    unsigned ldsByteAddr)
{
  unsigned long long ga = (unsigned long long)(size_t)gptr;
  u32x4 g0;
  g0[0] = 1u;                                        // count=1, user descriptor
  g0[1] = ldsByteAddr;                               // lds_addr
  g0[2] = (unsigned)(ga & 0xffffffffu);              // global_addr[31:0]
  g0[3] = (unsigned)((ga >> 32) & 0x01ffffffu)       // global_addr[56:32]
          | (2u << 30);                              // type = 2 ("image")
  i32x8 g1;
  g1[0] = (int)((1u << 16)        // data_size = 2 bytes
              | (1u << 20)        // pad_enable
              | (3u << 22)        // pad_interval: 16 DWORDs (64B) per row
              | (3u << 25));      // pad_amount: 4 DWORDs (16B)
  g1[1] = (int)((strideElems & 0xffffu) << 16);                  // tensor_dim0 lo16
  g1[2] = (int)(((strideElems >> 16) & 0xffffu)                  // tensor_dim0 hi16
              | ((rows & 0xffffu) << 16));                       // tensor_dim1 lo16
  g1[3] = (int)(((rows >> 16) & 0xffffu) | (32u << 16));         // tile_dim0 = 32
  g1[4] = (int)(rows & 0xffffu);                                 // tile_dim1 = rows
  g1[5] = (int)strideElems;                                      // dim0_stride lo32
  g1[6] = 0;                                                     // dim0_stride hi16
  g1[7] = 0;
  i32x4 z = {0, 0, 0, 0};
#if __clang_major__ >= 23
  i32x8 z8 = {0, 0, 0, 0, 0, 0, 0, 0};
  __builtin_amdgcn_tensor_load_to_lds(g0, g1, z, z, z8, 0);
#else
  __builtin_amdgcn_tensor_load_to_lds(g0, g1, z, z, 0);
#endif
}

// ---------------------------------------------------------------------------
// LayerNorm over last dim (512). One wave per row, lane owns 16 contiguous
// elements: vector loads in, one 32B bf16 store (or 4x float4) out.
// ---------------------------------------------------------------------------
__global__ __launch_bounds__(256) void ln_kernel(
    const float* __restrict__ x, const float* __restrict__ g,
    const float* __restrict__ bsh, unsigned short* __restrict__ outb,
    float* __restrict__ outf, int M)
{
  int row  = blockIdx.x * 8 + (threadIdx.x >> 5);
  int lane = threadIdx.x & 31;
  if (row >= M) return;
  const float* xr = x + (size_t)row * DM + lane * 16;
  float v[16];
  float s = 0.f;
#pragma unroll
  for (int i = 0; i < 16; i += 4) {
    float4 t = *(const float4*)(xr + i);
    v[i] = t.x; v[i+1] = t.y; v[i+2] = t.z; v[i+3] = t.w;
    s += t.x + t.y + t.z + t.w;
  }
#pragma unroll
  for (int o = 16; o > 0; o >>= 1) s += __shfl_xor(s, o, 32);
  float mu = s * (1.f / 512.f);
  float q = 0.f;
#pragma unroll
  for (int i = 0; i < 16; ++i) { float d = v[i] - mu; q += d * d; }
#pragma unroll
  for (int o = 16; o > 0; o >>= 1) q += __shfl_xor(q, o, 32);
  float inv = rsqrtf(q * (1.f / 512.f) + 1e-5f);
  const float* gp = g + lane * 16;
  const float* bp = bsh + lane * 16;
  if (outb) {
    us16 w;
#pragma unroll
    for (int i = 0; i < 16; ++i) w[i] = f2bf((v[i] - mu) * inv * gp[i] + bp[i]);
    *(us16*)(outb + (size_t)row * DM + lane * 16) = w;
  } else {
#pragma unroll
    for (int i = 0; i < 16; i += 4) {
      float4 o;
      o.x = (v[i]   - mu) * inv * gp[i]   + bp[i];
      o.y = (v[i+1] - mu) * inv * gp[i+1] + bp[i+1];
      o.z = (v[i+2] - mu) * inv * gp[i+2] + bp[i+2];
      o.w = (v[i+3] - mu) * inv * gp[i+3] + bp[i+3];
      *(float4*)(outf + (size_t)row * DM + lane * 16 + i) = o;
    }
  }
}

// ---------------------------------------------------------------------------
// One-time weight transform: W_f32[K][N] -> Wt_bf16[N][K] (tiled transpose).
// ---------------------------------------------------------------------------
__global__ __launch_bounds__(256) void tcast_kernel(
    const float* __restrict__ W, unsigned short* __restrict__ Wt, int K, int N)
{
  __shared__ unsigned short tile[32][33];
  int n0 = blockIdx.x * 32, k0 = blockIdx.y * 32;
  int lx = threadIdx.x & 31, ly = threadIdx.x >> 5;   // 32 x 8
#pragma unroll
  for (int i = 0; i < 4; ++i) {
    int k = ly + i * 8;
    tile[k][lx] = f2bf(W[(size_t)(k0 + k) * N + n0 + lx]);
  }
  __syncthreads();
#pragma unroll
  for (int i = 0; i < 4; ++i) {
    int n = ly + i * 8;
    Wt[(size_t)(n0 + n) * K + k0 + lx] = tile[lx][n];
  }
}

// ---------------------------------------------------------------------------
// Generic GEMM: C[M,N] = epi(A_bf16[M,K] @ Wt_bf16[N][K]^T + bias)
// Block tile 128x64, 8 waves (4x2), wave tile 32x32 (4 WMMA accumulators).
// Both tiles staged by TDM, double-buffered; 80B LDS rows via TDM padding.
// epi: 0=f32 store, 1=SiLU->bf16, 2=res+scale*acc->f32, 3=bf16,
//      4=dual bf16 (acc+eb1, acc+eb2), 5=bf16 transposed [N][M] store
// ---------------------------------------------------------------------------
#define BM 128
#define BN 64
#define BK 32
#define LSTR 40   // LDS row stride (ushorts): 80B rows (TDM pad), 16B aligned

__global__ __launch_bounds__(256) void gemm_bf16(
    const unsigned short* __restrict__ A, const unsigned short* __restrict__ Wt,
    const float* __restrict__ bias, int M, int K, int N,
    int epi, float scale, const float* __restrict__ res,
    float* __restrict__ outf, unsigned short* __restrict__ outb,
    unsigned short* __restrict__ outb2,
    const float* __restrict__ eb1, const float* __restrict__ eb2)
{
  __shared__ __attribute__((aligned(32))) unsigned short As[2][BM][LSTR];
  __shared__ __attribute__((aligned(32))) unsigned short Bs[2][BN][LSTR];
  const int tid  = threadIdx.x;
  const int lane = tid & 31;
  const int wave = tid >> 5;
  const int wm = wave & 3, wn = wave >> 2;
  const int half = lane >> 4, l15 = lane & 15;
  const int bm0 = blockIdx.y * BM;
  const int bn0 = blockIdx.x * BN;
  v8f acc[2][2] = {};

  const int nk = K / BK;
  if (wave == 0) {   // prologue: DMA first tiles into buffer 0
    tdm_load_tile(A  + (size_t)bm0 * K, BM, (unsigned)K,
                  (unsigned)(size_t)&As[0][0][0]);
    tdm_load_tile(Wt + (size_t)bn0 * K, BN, (unsigned)K,
                  (unsigned)(size_t)&Bs[0][0][0]);
    __builtin_amdgcn_s_wait_tensorcnt(0);
  }
  __syncthreads();

  for (int ik = 0; ik < nk; ++ik) {
    const int cur = ik & 1;
    if (ik + 1 < nk && wave == 0) {   // async-prefetch next tiles
      int k1 = (ik + 1) * BK;
      tdm_load_tile(A  + (size_t)bm0 * K + k1, BM, (unsigned)K,
                    (unsigned)(size_t)&As[1 - cur][0][0]);
      tdm_load_tile(Wt + (size_t)bn0 * K + k1, BN, (unsigned)K,
                    (unsigned)(size_t)&Bs[1 - cur][0][0]);
    }

    v16bf af[2], bg[2];
#pragma unroll
    for (int t = 0; t < 2; ++t) {
      int rb = wm * 32 + t * 16 + l15;
      af[t] = frag2(&As[cur][rb][half * 8], &As[cur][rb][16 + half * 8]);
      int nb = wn * 32 + t * 16 + l15;
      bg[t] = frag2(&Bs[cur][nb][half * 8], &Bs[cur][nb][16 + half * 8]);
    }
    acc[0][0] = __builtin_amdgcn_wmma_f32_16x16x32_bf16(false, af[0], false, bg[0], (short)0, acc[0][0], false, false);
    acc[0][1] = __builtin_amdgcn_wmma_f32_16x16x32_bf16(false, af[0], false, bg[1], (short)0, acc[0][1], false, false);
    acc[1][0] = __builtin_amdgcn_wmma_f32_16x16x32_bf16(false, af[1], false, bg[0], (short)0, acc[1][0], false, false);
    acc[1][1] = __builtin_amdgcn_wmma_f32_16x16x32_bf16(false, af[1], false, bg[1], (short)0, acc[1][1], false, false);

    if (wave == 0) __builtin_amdgcn_s_wait_tensorcnt(0);
    __syncthreads();   // next buffer published; current buffer free for DMA
  }

  // Epilogue: C layout VGPR e -> M = e + 8*half, N = lane&15.
#pragma unroll
  for (int ti = 0; ti < 2; ++ti)
#pragma unroll
    for (int tj = 0; tj < 2; ++tj)
#pragma unroll
      for (int e = 0; e < 8; ++e) {
        int gr = bm0 + wm * 32 + ti * 16 + e + half * 8;
        int gc = bn0 + wn * 32 + tj * 16 + l15;
        float v = acc[ti][tj][e];
        if (bias) v += bias[gc];
        size_t o = (size_t)gr * N + gc;
        if (epi == 0)      outf[o] = v;
        else if (epi == 1) outb[o] = f2bf(v * sigm(v));
        else if (epi == 2) outf[o] = res[o] + scale * v;
        else if (epi == 3) outb[o] = f2bf(v);
        else if (epi == 5) outb[(size_t)gc * M + gr] = f2bf(v);  // transposed
        else { outb[o] = f2bf(v + eb1[gc]); outb2[o] = f2bf(v + eb2[gc]); }
      }
}

// ---------------------------------------------------------------------------
// Attention: one block per (b, h, 16-row query tile).
// Content scores via WMMA (K fragments = contiguous 32B global vectors);
// relative-shift positional term via exact pad/reshape index algebra;
// softmax in LDS; attn @ V via WMMA against a pre-transposed V buffer.
// ---------------------------------------------------------------------------
__global__ __launch_bounds__(256) void attn_kernel(
    const unsigned short* __restrict__ qu, const unsigned short* __restrict__ qv,
    const unsigned short* __restrict__ kb, const unsigned short* __restrict__ vT,
    const unsigned short* __restrict__ pp, const int* __restrict__ mask,
    unsigned short* __restrict__ ao)
{
  __shared__ __attribute__((aligned(32))) unsigned short sc[16][TT];   // 32 KB
  __shared__ __attribute__((aligned(32))) unsigned short squ[16][DKH];
  __shared__ __attribute__((aligned(32))) unsigned short sqv[17][DKH]; // +1 guard row
  __shared__ float red[16][16];
  const int tid  = threadIdx.x;
  const int lane = tid & 31, wave = tid >> 5;
  const int half = lane >> 4, l15 = lane & 15;
  const int bid = blockIdx.x;
  const int rt = bid & 63;
  const int h  = (bid >> 6) & 7;
  const int b  = bid >> 9;
  const int r0 = rt * 16;

  for (int i = tid; i < 16 * DKH; i += 256)
    squ[i >> 6][i & 63] = qu[((size_t)(b * TT + r0 + (i >> 6))) * DM + h * DKH + (i & 63)];
  for (int i = tid; i < 17 * DKH; i += 256) {
    int rr = i >> 6, r = r0 + rr;
    sqv[rr][i & 63] = (r < TT)
        ? qv[((size_t)(b * TT + r)) * DM + h * DKH + (i & 63)]
        : (unsigned short)0;
  }
  __syncthreads();

  // A fragments (q+u tile) are loop invariant: 2x ds_load_b128 each.
  v16bf a0 = frag2(&squ[l15][half * 16],      &squ[l15][half * 16 + 8]);
  v16bf a1 = frag2(&squ[l15][32 + half * 16], &squ[l15][40 + half * 16]);

  // Phase 1: scores. 8 waves x 8 key tiles of 16 = 1024 columns.
  for (int it = 0; it < 8; ++it) {
    int s0 = (wave + 8 * it) * 16;
    const unsigned short* kptr = kb + ((size_t)(b * TT + s0 + l15)) * DM + h * DKH;
    v16bf b0 = frag1(kptr + half * 16);        // 2x global_load_b128
    v16bf b1 = frag1(kptr + 32 + half * 16);
    v8f cs = {};
    cs = __builtin_amdgcn_wmma_f32_16x16x32_bf16(false, a0, false, b0, (short)0, cs, false, false);
    cs = __builtin_amdgcn_wmma_f32_16x16x32_bf16(false, a1, false, b1, (short)0, cs, false, false);
#pragma unroll 1
    for (int e = 0; e < 8; ++e) {
      int rl = e + half * 8;
      int r  = r0 + rl;
      int cc = s0 + l15;
      // _relative_shift: pad-left / reshape / drop-row index algebra.
      float ps = 0.f;
      int qrow = -1, j = 0;
      if (cc <= r)          { qrow = rl;     j = (TT - 1) + cc - r; }
      else if (cc > r + 1)  { qrow = rl + 1; j = cc - r - 2; }
      if (qrow >= 0) {
        const unsigned short* pr = pp + (size_t)j * DM + h * DKH;
        const unsigned short* qr = &sqv[qrow][0];
#pragma unroll 1
        for (int kk = 0; kk < DKH; kk += 8) {
          us8 qa = *(const us8*)(qr + kk);   // ds_load_b128
          us8 pa = *(const us8*)(pr + kk);   // global_load_b128
#pragma unroll
          for (int e2 = 0; e2 < 8; ++e2) ps += bf2f(qa[e2]) * bf2f(pa[e2]);
        }
      } // cc == r+1 -> zero contribution (the pad column)
      float val = (cs[e] + ps) * 0.125f;   // 1/sqrt(DK)
      if (mask[b * TT + cc] == 0) val = -3.0e38f;
      sc[rl][cc] = f2bf(val);
    }
  }
  __syncthreads();

  // Phase 2: softmax over 1024 columns. 16 threads per row.
  {
    int row = tid >> 4, sub = tid & 15;
    float m = -3.0e38f;
    for (int j = 0; j < TT / 16; ++j) m = fmaxf(m, bf2f(sc[row][sub + 16 * j]));
    red[row][sub] = m;
    __syncthreads();
    m = -3.0e38f;
    for (int k2 = 0; k2 < 16; ++k2) m = fmaxf(m, red[row][k2]);
    __syncthreads();
    float s = 0.f;
    for (int j = 0; j < TT / 16; ++j) {
      int cc = sub + 16 * j;
      float v = __expf(bf2f(sc[row][cc]) - m);
      sc[row][cc] = f2bf(v);
      s += v;
    }
    red[row][sub] = s;
    __syncthreads();
    s = 0.f;
    for (int k2 = 0; k2 < 16; ++k2) s += red[row][k2];
    float inv = 1.f / s;
    for (int j = 0; j < TT / 16; ++j) {
      int cc = sub + 16 * j;
      sc[row][cc] = f2bf(bf2f(sc[row][cc]) * inv);
    }
  }
  __syncthreads();

  // Phase 3: out = attn @ V (V pre-transposed: vT[d][b*T + s]).
  if (wave < 4) {
    const unsigned short* vptr =
        vT + ((size_t)(h * DKH + wave * 16 + l15)) * (size_t)(NBAT * TT) + (size_t)b * TT;
    v8f o = {};
    for (int ks = 0; ks < TT; ks += 32) {
      v16bf a  = frag2(&sc[l15][ks + half * 16], &sc[l15][ks + half * 16 + 8]);
      v16bf bb = frag1(vptr + ks + half * 16);   // 2x global_load_b128
      o = __builtin_amdgcn_wmma_f32_16x16x32_bf16(false, a, false, bb, (short)0, o, false, false);
    }
#pragma unroll
    for (int e = 0; e < 8; ++e) {
      int rl = e + half * 8;
      ao[((size_t)(b * TT + r0 + rl)) * DM + h * DKH + wave * 16 + l15] = f2bf(o[e]);
    }
  }
}

// ---------------------------------------------------------------------------
// Elementwise helpers
// ---------------------------------------------------------------------------
__global__ __launch_bounds__(256) void cast_bf16_kernel(
    const float* __restrict__ in, unsigned short* __restrict__ out, int n)
{
  int i = blockIdx.x * 256 + threadIdx.x;
  if (i < n) out[i] = f2bf(in[i]);
}

__global__ __launch_bounds__(256) void glu_kernel(
    const float* __restrict__ h, float* __restrict__ out, int m)
{
  int i = blockIdx.x * 256 + threadIdx.x;
  if (i >= m * DM) return;
  int r = i >> 9, c = i & 511;
  float a  = h[(size_t)r * 1024 + c];
  float bb = h[(size_t)r * 1024 + 512 + c];
  out[i] = a * sigm(bb);
}

__global__ __launch_bounds__(256) void dwconv_kernel(
    const float* __restrict__ in, const float* __restrict__ w,
    const float* __restrict__ wb, const float* __restrict__ bng,
    const float* __restrict__ bnb, const float* __restrict__ bnm,
    const float* __restrict__ bnv, unsigned short* __restrict__ out)
{
  int idx = blockIdx.x * 256 + threadIdx.x;
  if (idx >= NBAT * TT * DM) return;
  int ch = idx & (DM - 1);
  int t  = (idx >> 9) & (TT - 1);
  int b  = idx >> 19;
  float a = 0.f;
#pragma unroll 1
  for (int k = 0; k < 31; ++k) {
    int t2 = t + k - 15;
    if (t2 >= 0 && t2 < TT)
      a += in[((size_t)(b * TT + t2)) * DM + ch] * w[ch * 31 + k];
  }
  a += wb[ch];
  a = (a - bnm[ch]) * rsqrtf(bnv[ch] + 1e-5f) * bng[ch] + bnb[ch];
  a = a * sigm(a);
  out[idx] = f2bf(a);
}

// ---------------------------------------------------------------------------
// Host orchestration
// ---------------------------------------------------------------------------
extern "C" void kernel_launch(void* const* d_in, const int* in_sizes, int n_in,
                              void* d_out, int out_size, void* d_ws, size_t ws_size,
                              hipStream_t stream) {
  (void)in_sizes; (void)n_in; (void)out_size; (void)ws_size;
  const float* x        = (const float*)d_in[0];
  const float* posenc   = (const float*)d_in[1];
  const int*   mask     = (const int*)d_in[2];
  const float* ff1_ln_g = (const float*)d_in[3];
  const float* ff1_ln_b = (const float*)d_in[4];
  const float* ff1_w1   = (const float*)d_in[5];
  const float* ff1_b1   = (const float*)d_in[6];
  const float* ff1_w2   = (const float*)d_in[7];
  const float* ff1_b2   = (const float*)d_in[8];
  const float* ff2_ln_g = (const float*)d_in[9];
  const float* ff2_ln_b = (const float*)d_in[10];
  const float* ff2_w1   = (const float*)d_in[11];
  const float* ff2_b1   = (const float*)d_in[12];
  const float* ff2_w2   = (const float*)d_in[13];
  const float* ff2_b2   = (const float*)d_in[14];
  const float* attn_ln_g= (const float*)d_in[15];
  const float* attn_ln_b= (const float*)d_in[16];
  const float* wq       = (const float*)d_in[17];
  const float* wk       = (const float*)d_in[18];
  const float* wv       = (const float*)d_in[19];
  const float* wpos     = (const float*)d_in[20];
  const float* wout     = (const float*)d_in[21];
  const float* bq       = (const float*)d_in[22];
  const float* bk       = (const float*)d_in[23];
  const float* bv       = (const float*)d_in[24];
  const float* bout     = (const float*)d_in[25];
  const float* pbu      = (const float*)d_in[26];
  const float* pbv      = (const float*)d_in[27];
  const float* conv_ln_g= (const float*)d_in[28];
  const float* conv_ln_b= (const float*)d_in[29];
  const float* pw1_w    = (const float*)d_in[30];
  const float* pw1_b    = (const float*)d_in[31];
  const float* dw_w     = (const float*)d_in[32];
  const float* dw_b     = (const float*)d_in[33];
  const float* bn_g     = (const float*)d_in[34];
  const float* bn_b     = (const float*)d_in[35];
  const float* bn_mean  = (const float*)d_in[36];
  const float* bn_var   = (const float*)d_in[37];
  const float* pw2_w    = (const float*)d_in[38];
  const float* pw2_b    = (const float*)d_in[39];
  const float* fin_g    = (const float*)d_in[40];
  const float* fin_b    = (const float*)d_in[41];

  const int M = NBAT * TT;  // 8192 rows

  char* wsp = (char*)d_ws;
  size_t off = 0;
  auto alloc = [&](size_t bytes) -> char* {
    char* p = wsp + off;
    off += (bytes + 255) & ~(size_t)255;
    return p;
  };
  float* x1 = (float*)alloc((size_t)M * DM * 4);
  float* x2 = (float*)alloc((size_t)M * DM * 4);
  float* x3 = (float*)alloc((size_t)M * DM * 4);
  unsigned short* lnb   = (unsigned short*)alloc((size_t)M * DM * 2);
  unsigned short* hb    = (unsigned short*)alloc((size_t)M * 2048 * 2);
  unsigned short* qub   = (unsigned short*)alloc((size_t)M * DM * 2);
  unsigned short* qvb   = (unsigned short*)alloc((size_t)M * DM * 2);
  unsigned short* kbuf  = (unsigned short*)alloc((size_t)M * DM * 2);
  unsigned short* vtb   = (unsigned short*)alloc((size_t)DM * M * 2);  // V transposed [D][M]
  unsigned short* posbf = (unsigned short*)alloc((size_t)TT * DM * 2);
  unsigned short* posp  = (unsigned short*)alloc((size_t)TT * DM * 2);
  unsigned short* attnb = (unsigned short*)alloc((size_t)M * DM * 2);
  float* pw1buf = (float*)alloc((size_t)M * 1024 * 4);
  float* glubuf = (float*)alloc((size_t)M * DM * 4);
  unsigned short* convb = (unsigned short*)alloc((size_t)M * DM * 2);
  // Transposed bf16 weights Wt[N][K]:
  unsigned short* wt_ff1_1 = (unsigned short*)alloc((size_t)2048 * 512 * 2);
  unsigned short* wt_ff1_2 = (unsigned short*)alloc((size_t)512 * 2048 * 2);
  unsigned short* wt_ff2_1 = (unsigned short*)alloc((size_t)2048 * 512 * 2);
  unsigned short* wt_ff2_2 = (unsigned short*)alloc((size_t)512 * 2048 * 2);
  unsigned short* wt_q     = (unsigned short*)alloc((size_t)512 * 512 * 2);
  unsigned short* wt_k     = (unsigned short*)alloc((size_t)512 * 512 * 2);
  unsigned short* wt_v     = (unsigned short*)alloc((size_t)512 * 512 * 2);
  unsigned short* wt_pos   = (unsigned short*)alloc((size_t)512 * 512 * 2);
  unsigned short* wt_out   = (unsigned short*)alloc((size_t)512 * 512 * 2);
  unsigned short* wt_pw1   = (unsigned short*)alloc((size_t)1024 * 512 * 2);
  unsigned short* wt_pw2   = (unsigned short*)alloc((size_t)512 * 512 * 2);

  dim3 blk(256);

  // One-time weight transpose+cast (K, N of the original row-major weight).
  tcast_kernel<<<dim3(2048/32, 512/32),  blk, 0, stream>>>(ff1_w1, wt_ff1_1, 512, 2048);
  tcast_kernel<<<dim3(512/32,  2048/32), blk, 0, stream>>>(ff1_w2, wt_ff1_2, 2048, 512);
  tcast_kernel<<<dim3(2048/32, 512/32),  blk, 0, stream>>>(ff2_w1, wt_ff2_1, 512, 2048);
  tcast_kernel<<<dim3(512/32,  2048/32), blk, 0, stream>>>(ff2_w2, wt_ff2_2, 2048, 512);
  tcast_kernel<<<dim3(512/32,  512/32),  blk, 0, stream>>>(wq,    wt_q,   512, 512);
  tcast_kernel<<<dim3(512/32,  512/32),  blk, 0, stream>>>(wk,    wt_k,   512, 512);
  tcast_kernel<<<dim3(512/32,  512/32),  blk, 0, stream>>>(wv,    wt_v,   512, 512);
  tcast_kernel<<<dim3(512/32,  512/32),  blk, 0, stream>>>(wpos,  wt_pos, 512, 512);
  tcast_kernel<<<dim3(512/32,  512/32),  blk, 0, stream>>>(wout,  wt_out, 512, 512);
  tcast_kernel<<<dim3(1024/32, 512/32),  blk, 0, stream>>>(pw1_w, wt_pw1, 512, 1024);
  tcast_kernel<<<dim3(512/32,  512/32),  blk, 0, stream>>>(pw2_w, wt_pw2, 512, 512);

  // x = x + 0.5 * ff1(x)
  ln_kernel<<<M / 8, blk, 0, stream>>>(x, ff1_ln_g, ff1_ln_b, lnb, nullptr, M);
  gemm_bf16<<<dim3(2048 / BN, M / BM), blk, 0, stream>>>(
      lnb, wt_ff1_1, ff1_b1, M, 512, 2048, 1, 0.f, nullptr, nullptr, hb, nullptr, nullptr, nullptr);
  gemm_bf16<<<dim3(512 / BN, M / BM), blk, 0, stream>>>(
      hb, wt_ff1_2, ff1_b2, M, 2048, 512, 2, 0.5f, x, x1, nullptr, nullptr, nullptr, nullptr);

  // attention
  ln_kernel<<<M / 8, blk, 0, stream>>>(x1, attn_ln_g, attn_ln_b, lnb, nullptr, M);
  gemm_bf16<<<dim3(512 / BN, M / BM), blk, 0, stream>>>(
      lnb, wt_q, bq, M, 512, 512, 4, 0.f, nullptr, nullptr, qub, qvb, pbu, pbv);
  gemm_bf16<<<dim3(512 / BN, M / BM), blk, 0, stream>>>(
      lnb, wt_k, bk, M, 512, 512, 3, 0.f, nullptr, nullptr, kbuf, nullptr, nullptr, nullptr);
  gemm_bf16<<<dim3(512 / BN, M / BM), blk, 0, stream>>>(
      lnb, wt_v, bv, M, 512, 512, 5, 0.f, nullptr, nullptr, vtb, nullptr, nullptr, nullptr);
  cast_bf16_kernel<<<(TT * DM) / 256, blk, 0, stream>>>(posenc, posbf, TT * DM);
  gemm_bf16<<<dim3(512 / BN, TT / BM), blk, 0, stream>>>(
      posbf, wt_pos, nullptr, TT, 512, 512, 3, 0.f, nullptr, nullptr, posp, nullptr, nullptr, nullptr);
  attn_kernel<<<NBAT * NH * (TT / 16), blk, 0, stream>>>(
      qub, qvb, kbuf, vtb, posp, mask, attnb);
  gemm_bf16<<<dim3(512 / BN, M / BM), blk, 0, stream>>>(
      attnb, wt_out, bout, M, 512, 512, 2, 1.0f, x1, x2, nullptr, nullptr, nullptr, nullptr);

  // conv module
  ln_kernel<<<M / 8, blk, 0, stream>>>(x2, conv_ln_g, conv_ln_b, lnb, nullptr, M);
  gemm_bf16<<<dim3(1024 / BN, M / BM), blk, 0, stream>>>(
      lnb, wt_pw1, pw1_b, M, 512, 1024, 0, 0.f, nullptr, pw1buf, nullptr, nullptr, nullptr, nullptr);
  glu_kernel<<<(M * DM) / 256, blk, 0, stream>>>(pw1buf, glubuf, M);
  dwconv_kernel<<<(M * DM) / 256, blk, 0, stream>>>(
      glubuf, dw_w, dw_b, bn_g, bn_b, bn_mean, bn_var, convb);
  gemm_bf16<<<dim3(512 / BN, M / BM), blk, 0, stream>>>(
      convb, wt_pw2, pw2_b, M, 512, 512, 2, 1.0f, x2, x3, nullptr, nullptr, nullptr, nullptr);

  // x = x + 0.5 * ff2(x)
  ln_kernel<<<M / 8, blk, 0, stream>>>(x3, ff2_ln_g, ff2_ln_b, lnb, nullptr, M);
  gemm_bf16<<<dim3(2048 / BN, M / BM), blk, 0, stream>>>(
      lnb, wt_ff2_1, ff2_b1, M, 512, 2048, 1, 0.f, nullptr, nullptr, hb, nullptr, nullptr, nullptr);
  gemm_bf16<<<dim3(512 / BN, M / BM), blk, 0, stream>>>(
      hb, wt_ff2_2, ff2_b2, M, 2048, 512, 2, 0.5f, x3, x1, nullptr, nullptr, nullptr, nullptr);

  // final LN -> d_out (fp32)
  ln_kernel<<<M / 8, blk, 0, stream>>>(x1, fin_g, fin_b, nullptr, (float*)d_out, M);
}
